// MegaBlocksMoeMLP_12747462934734
// MI455X (gfx1250) — matrix operand order repaired and checked by reading, not compiled, not run on hardware
//
#include <hip/hip_runtime.h>
#include <hip/hip_bf16.h>
#include <stdint.h>

// ---------------------------------------------------------------------------
// MoE MLP (gpt-oss swiglu, top-4 of 8 experts) for gfx1250 / MI455X.
// Dense per-expert bf16 WMMA GEMMs (v_wmma_f32_16x16x32_bf16), f32 accumulate.
// Block tile 32x256 (4 waves x [32x64]); K slabs staged in LDS with
// global_load_async_to_lds_b128 (ASYNCcnt) double buffering; fragments via
// ds_load_b128.
// Workspace (~64.2 MB):
//   xb   : bf16 x            4096*1024
//   w1t  : bf16 w1^T (N,K)   8*2048*1024
//   w2t  : bf16 w2^T (N,K)   8*1024*1024
//   act  : bf16 activations  4096*1024   (reused per expert)
//   comb : f32 combine       8*4096
// ---------------------------------------------------------------------------

#define T_TOK 4096
#define HDIM  1024
#define IDIM  1024
#define NEXP  8

#define ROWP     40                       // LDS row pitch in bf16 elements (80B)
#define A_BYTES  (32 * ROWP * 2)          // 2560 B per A slab
#define B_BYTES  (256 * ROWP * 2)         // 20480 B per B slab
#define LDS_BYTES (2 * A_BYTES + 2 * B_BYTES)   // 46080 B

typedef __attribute__((ext_vector_type(16))) __bf16 v16bf;
typedef __attribute__((ext_vector_type(8)))  float  v8f;

union BfFrag {
    uint4 q[2];   // 32B = 16 bf16
    v16bf v;
};

#define WMMA_BF16(a, b, c) \
    __builtin_amdgcn_wmma_f32_16x16x32_bf16(false, (a), false, (b), (short)0, (c), false, false)

__device__ __forceinline__ uint16_t f32_to_bf16(float f) {
    uint32_t u = __float_as_uint(f);
    uint32_t r = u + 0x7FFFu + ((u >> 16) & 1u);   // round-to-nearest-even
    return (uint16_t)(r >> 16);
}

// Async global -> LDS, 16B per lane. lds_off is a byte offset from the
// workgroup LDS base (dynamic-shared segment assumed at offset 0).
__device__ __forceinline__ void async_ld_b128(uint32_t lds_off, const void* gptr) {
    asm volatile("global_load_async_to_lds_b128 %0, %1, off"
                 :: "v"(lds_off), "v"(gptr) : "memory");
}
#define WAIT_ASYNC(n) asm volatile("s_wait_asynccnt %0" :: "i"(n) : "memory")

// ---------------------------- f32 -> bf16 copy -----------------------------
__global__ void k_cvt_bf16(const float* __restrict__ in,
                           uint16_t* __restrict__ out, int n) {
    int i = blockIdx.x * blockDim.x + threadIdx.x;
    if (i < n) out[i] = f32_to_bf16(in[i]);
}

// ------------------- per-expert tiled transpose f32->bf16 ------------------
// in: (E, R, C) f32 row-major  ->  out: (E, C, R) bf16 row-major
__global__ void k_transpose_bf16(const float* __restrict__ in,
                                 uint16_t* __restrict__ out, int R, int C) {
    __shared__ float tile[32][33];
    int e = blockIdx.z;
    const float* src = in + (size_t)e * R * C;
    uint16_t*    dst = out + (size_t)e * R * C;
    int c0 = blockIdx.x * 32, r0 = blockIdx.y * 32;
    int tx = threadIdx.x, ty = threadIdx.y;            // 32 x 8
    #pragma unroll
    for (int i = ty; i < 32; i += 8)
        tile[i][tx] = src[(size_t)(r0 + i) * C + (c0 + tx)];
    __syncthreads();
    #pragma unroll
    for (int i = ty; i < 32; i += 8)
        dst[(size_t)(c0 + i) * R + (r0 + tx)] = f32_to_bf16(tile[tx][i]);
}

// ------------------------------- router ------------------------------------
__global__ void k_router(const float* __restrict__ x,
                         const float* __restrict__ rw,
                         const float* __restrict__ rb,
                         float* __restrict__ ew_out,     // (T,4)
                         float* __restrict__ combine) {  // (E,T)
    int wave = threadIdx.x >> 5;
    int lane = threadIdx.x & 31;
    int t = blockIdx.x * (blockDim.x >> 5) + wave;
    if (t >= T_TOK) return;
    const float* xr = x + (size_t)t * HDIM;
    float v[NEXP];
    #pragma unroll
    for (int e = 0; e < NEXP; ++e) {
        const float* wr = rw + (size_t)e * HDIM;
        float acc = 0.f;
        for (int h = lane; h < HDIM; h += 32) acc += xr[h] * wr[h];
        #pragma unroll
        for (int s = 16; s > 0; s >>= 1) acc += __shfl_down(acc, s, 32);
        v[e] = acc + rb[e];   // valid on lane 0
    }
    if (lane == 0) {
        int   idx[4];
        float val[4];
        unsigned used = 0;
        #pragma unroll
        for (int j = 0; j < 4; ++j) {
            int bi = -1; float bv = -3.4e38f;
            #pragma unroll
            for (int e = 0; e < NEXP; ++e)
                if (!((used >> e) & 1) && v[e] > bv) { bv = v[e]; bi = e; }
            used |= 1u << bi; idx[j] = bi; val[j] = bv;
        }
        float m = val[0];
        float w[4], s = 0.f;
        #pragma unroll
        for (int j = 0; j < 4; ++j) { w[j] = expf(val[j] - m); s += w[j]; }
        float inv = 1.f / s;
        #pragma unroll
        for (int j = 0; j < 4; ++j) { w[j] *= inv; ew_out[t * 4 + j] = w[j]; }
        #pragma unroll
        for (int e = 0; e < NEXP; ++e) combine[(size_t)e * T_TOK + t] = 0.f;
        #pragma unroll
        for (int j = 0; j < 4; ++j) combine[(size_t)idx[j] * T_TOK + t] = w[j];
    }
}

// ---------------- shared GEMM core: LDS-staged, async-prefetched -----------
// aG = A global base at (m0, 0), row-major (*, K) bf16
// bG = B global base at (n0blk, 0), N-major (*, K) bf16
// Accumulates c[mi][j]: wave tile M=32 (2x16), N=64 (4x16).
__device__ __forceinline__ void async_copy_slab(const uint16_t* aRow,
                                                const uint16_t* bRow,
                                                int K, int bufA_b, int bufB_b,
                                                int tid) {
    int row = tid >> 2, chunk = tid & 3;          // 128 threads: 32 rows x 4 chunks
    uint32_t ldsA = (uint32_t)(bufA_b + (row * ROWP + chunk * 8) * 2);
    async_ld_b128(ldsA, aRow + (size_t)row * K + chunk * 8);
    #pragma unroll
    for (int i = 0; i < 8; ++i) {
        int nr = row + i * 32;                    // 256 B rows
        uint32_t ldsB = (uint32_t)(bufB_b + (nr * ROWP + chunk * 8) * 2);
        async_ld_b128(ldsB, bRow + (size_t)nr * K + chunk * 8);
    }
}

__device__ __forceinline__ void gemm_tiles(const uint16_t* aG, const uint16_t* bG,
                                           int K, uint16_t* smem, v8f c[2][4]) {
    int tid  = threadIdx.x;
    int wave = tid >> 5, lane = tid & 31;
    int hsel = lane >> 4, l16 = lane & 15;

    int nsteps = K / 32;
    async_copy_slab(aG, bG, K, /*bufA*/0, /*bufB*/2 * A_BYTES, tid);

    for (int s = 0; s < nsteps; ++s) {
        int cur = s & 1;
        bool pf = (s + 1 < nsteps);
        if (pf) {
            int nb = cur ^ 1;
            async_copy_slab(aG + (s + 1) * 32, bG + (s + 1) * 32, K,
                            nb * A_BYTES, 2 * A_BYTES + nb * B_BYTES, tid);
            WAIT_ASYNC(9);      // 9 in flight for next; cur slab complete
        } else {
            WAIT_ASYNC(0);
        }
        __syncthreads();

        const uint16_t* sA = smem + (cur * A_BYTES) / 2;
        const uint16_t* sB = smem + (2 * A_BYTES + cur * B_BYTES) / 2;

        BfFrag a0, a1, bf;
        {
            const uint16_t* p0 = sA + (l16 * ROWP + hsel * 8);
            a0.q[0] = *(const uint4*)(p0);
            a0.q[1] = *(const uint4*)(p0 + 16);
            const uint16_t* p1 = sA + ((l16 + 16) * ROWP + hsel * 8);
            a1.q[0] = *(const uint4*)(p1);
            a1.q[1] = *(const uint4*)(p1 + 16);
        }
        #pragma unroll
        for (int j = 0; j < 4; ++j) {
            int nr = wave * 64 + j * 16 + l16;
            const uint16_t* p = sB + (nr * ROWP + hsel * 8);
            bf.q[0] = *(const uint4*)(p);
            bf.q[1] = *(const uint4*)(p + 16);
            c[0][j] = WMMA_BF16(a0.v, bf.v, c[0][j]);
            c[1][j] = WMMA_BF16(a1.v, bf.v, c[1][j]);
        }
        __syncthreads();        // before this buffer is overwritten again
    }
}

// ----------------------- GEMM1 + fused SwiGLU (per expert) -----------------
__global__ __launch_bounds__(128)
void k_gemm1(const uint16_t* __restrict__ xb,
             const uint16_t* __restrict__ w1t,
             const float* __restrict__ w1_bias,
             uint16_t* __restrict__ act, int e) {
    extern __shared__ uint16_t smem[];
    const uint16_t* wB = w1t + (size_t)e * (2 * IDIM) * HDIM;
    const float*    b1 = w1_bias + (size_t)e * (2 * IDIM);
    int wave = threadIdx.x >> 5, lane = threadIdx.x & 31;
    int hsel = lane >> 4, l16 = lane & 15;
    int m0 = blockIdx.x * 32;
    int n0blk = blockIdx.y * 256;
    int n0 = n0blk + wave * 64;

    v8f c[2][4] = {{v8f{}, v8f{}, v8f{}, v8f{}}, {v8f{}, v8f{}, v8f{}, v8f{}}};
    gemm_tiles(xb + (size_t)m0 * HDIM, wB + (size_t)n0blk * HDIM, HDIM, smem, c);

    // Epilogue: D element (vgpr r, lane) -> M = m0 + mi*16 + r + 8*hsel, N = l16.
    // Even N = gate, odd N = up -> pair via shfl_xor(1), fused SwiGLU.
    #pragma unroll
    for (int mi = 0; mi < 2; ++mi) {
        int mBase = m0 + mi * 16 + hsel * 8;
        #pragma unroll
        for (int j = 0; j < 4; ++j) {
            int n = n0 + j * 16 + l16;
            float bias = b1[n];
            #pragma unroll
            for (int r = 0; r < 8; ++r) {
                float d = c[mi][j][r] + bias;
                float other = __shfl_xor(d, 1, 32);
                if ((l16 & 1) == 0) {
                    float gate = fminf(d, 7.0f);
                    float up   = fminf(fmaxf(other, -7.0f), 7.0f);
                    float glu  = gate / (1.0f + __expf(-1.702f * gate));
                    float a_   = (up + 1.0f) * glu;
                    act[(size_t)(mBase + r) * IDIM + (n >> 1)] = f32_to_bf16(a_);
                }
            }
        }
    }
}

// --------------------- GEMM2 + bias + combine scale (per expert) -----------
__global__ __launch_bounds__(128)
void k_gemm2(const uint16_t* __restrict__ act,
             const uint16_t* __restrict__ w2t,
             const float* __restrict__ w2_bias,
             const float* __restrict__ combine,
             float* __restrict__ out, int e, int accumulate) {
    extern __shared__ uint16_t smem[];
    const uint16_t* wB  = w2t + (size_t)e * HDIM * IDIM;
    const float*    b2  = w2_bias + (size_t)e * HDIM;
    const float*    cmb = combine + (size_t)e * T_TOK;
    int wave = threadIdx.x >> 5, lane = threadIdx.x & 31;
    int hsel = lane >> 4, l16 = lane & 15;
    int m0 = blockIdx.x * 32;
    int n0blk = blockIdx.y * 256;
    int n0 = n0blk + wave * 64;

    v8f c[2][4] = {{v8f{}, v8f{}, v8f{}, v8f{}}, {v8f{}, v8f{}, v8f{}, v8f{}}};
    gemm_tiles(act + (size_t)m0 * IDIM, wB + (size_t)n0blk * IDIM, IDIM, smem, c);

    #pragma unroll
    for (int mi = 0; mi < 2; ++mi) {
        int mBase = m0 + mi * 16 + hsel * 8;
        #pragma unroll
        for (int j = 0; j < 4; ++j) {
            int n = n0 + j * 16 + l16;
            float bias = b2[n];
            #pragma unroll
            for (int r = 0; r < 8; ++r) {
                int m = mBase + r;
                float val = (c[mi][j][r] + bias) * cmb[m];
                size_t o = (size_t)m * HDIM + n;
                if (accumulate) out[o] += val;
                else            out[o]  = val;
            }
        }
    }
}

// ---------------------------------------------------------------------------
extern "C" void kernel_launch(void* const* d_in, const int* in_sizes, int n_in,
                              void* d_out, int out_size, void* d_ws, size_t ws_size,
                              hipStream_t stream) {
    const float* x   = (const float*)d_in[0];   // (2,2048,1024)
    const float* rw  = (const float*)d_in[1];   // (8,1024)
    const float* rb  = (const float*)d_in[2];   // (8)
    const float* w1  = (const float*)d_in[3];   // (8,1024,2048)
    const float* b1  = (const float*)d_in[4];   // (8,2048)
    const float* w2  = (const float*)d_in[5];   // (8,1024,1024)
    const float* b2  = (const float*)d_in[6];   // (8,1024)

    float* out = (float*)d_out;                       // T*H
    float* ew  = out + (size_t)T_TOK * HDIM;          // T*4

    uint16_t* xb   = (uint16_t*)d_ws;
    uint16_t* w1t  = xb  + (size_t)T_TOK * HDIM;
    uint16_t* w2t  = w1t + (size_t)NEXP * 2 * IDIM * HDIM;
    uint16_t* act  = w2t + (size_t)NEXP * HDIM * IDIM;
    float*    comb = (float*)(act + (size_t)T_TOK * IDIM);

    {
        int n = T_TOK * HDIM;
        k_cvt_bf16<<<(n + 255) / 256, 256, 0, stream>>>(x, xb, n);
    }
    k_transpose_bf16<<<dim3(2 * IDIM / 32, HDIM / 32, NEXP), dim3(32, 8), 0, stream>>>(
        w1, w1t, HDIM, 2 * IDIM);
    k_transpose_bf16<<<dim3(HDIM / 32, IDIM / 32, NEXP), dim3(32, 8), 0, stream>>>(
        w2, w2t, IDIM, HDIM);

    k_router<<<T_TOK / 8, 256, 0, stream>>>(x, rw, rb, ew, comb);

    for (int e = 0; e < NEXP; ++e) {
        k_gemm1<<<dim3(T_TOK / 32, (2 * IDIM) / 256), 128, LDS_BYTES, stream>>>(
            xb, w1t, b1, act, e);
        k_gemm2<<<dim3(T_TOK / 32, HDIM / 256), 128, LDS_BYTES, stream>>>(
            act, w2t, b2, comb, out, e, e > 0);
    }
}
